// CNNPolicyHead_34213709480444
// MI455X (gfx1250) — matrix operand order repaired and checked by reading, not compiled, not run on hardware
//
#include <hip/hip_runtime.h>
#include <hip/hip_bf16.h>
#include <math.h>

typedef __attribute__((ext_vector_type(16))) __bf16 v16bf;
typedef __attribute__((ext_vector_type(8)))  __bf16 v8bf;
typedef __attribute__((ext_vector_type(8)))  float  v8f;

#define HW      361                 // 19*19
#define CIN     384
#define CMB     96                  // 48 (p) + 48 (g) fused rows
#define MTILES  6                   // 96/16
#define NTILES  23                  // ceil(361/16) -> cols padded to 368
#define TILES   (MTILES * NTILES)   // 138
#define DPITCH  368                 // fp32 output row pitch (elements)
#define XPITCH  208                 // bf16 entries per position (192 data + 16 pad) = 416B
#define DSIZE   (CMB * DPITCH * 4)          // 141312 B
#define XOFF    DSIZE
#define XSIZE   (368 * XPITCH * 2)          // 153088 B
#define GPOFF   (XOFF + XSIZE)              // 294400 B
#define SMEM_TOTAL (GPOFF + (144 + 48 + 48 + 96) * 4)  // 295744 B < 320KB
#define NTHREADS 512                // 16 waves -> 4 waves per SIMD32
#define NWAVES   16

union AFrag { v16bf v; v8bf h[2]; };

// Stage one 192-channel half of x[n] into LDS as bf16, transposed to
// position-major so WMMA B-fragments are 32B-contiguous LDS reads.
// float4 (global_load_b128) reads: 192*361 floats = 17328 exact float4s.
__device__ __forceinline__ void stage_x(const float* __restrict__ xn, __bf16* X,
                                        int tid, int half) {
  const float* src = xn + half * (192 * HW);
  const float4* src4 = (const float4*)src;
  for (int f = tid; f < (192 * HW) / 4; f += NTHREADS) {
    const float4 v = src4[f];
    const float tmp[4] = {v.x, v.y, v.z, v.w};
    const int e0 = 4 * f;
#pragma unroll
    for (int j = 0; j < 4; ++j) {
      const int e   = e0 + j;
      const int c   = e / HW;
      const int pos = e - c * HW;
      X[pos * XPITCH + c] = (__bf16)tmp[j];
    }
    if (half == 0) __builtin_prefetch(src4 + (192 * HW) / 4 + f, 0, 0);
  }
}

// One K-half (192 channels = 6 x k32 steps) of the fused 96x361 GEMM.
// Each wave owns tiles t = wave, wave+16, ... ; accumulates through LDS D.
__device__ __forceinline__ void gemm_half(const __bf16* __restrict__ W,
                                          const __bf16* X, float* D,
                                          int wave, int lane, int half) {
  const int l15 = lane & 15;
  const int hi  = lane >> 4;
  for (int t = wave; t < TILES; t += NWAVES) {
    const int mt  = t / NTILES;
    const int nt  = t - mt * NTILES;
    const int col = nt * 16 + l15;     // output column (N)
    const int row = mt * 16 + l15;     // weight row (M)
    v8f c = {0.f, 0.f, 0.f, 0.f, 0.f, 0.f, 0.f, 0.f};
    if (half != 0) {
#pragma unroll
      for (int r = 0; r < 8; ++r)      // C/D layout: VGPR r -> M = r (+8 for hi lanes)
        c[r] = D[(mt * 16 + hi * 8 + r) * DPITCH + col];
    }
#pragma unroll
    for (int kk = 0; kk < 6; ++kk) {
      // A (16x32 bf16): lanes 0-15 hold K 0..15, lanes 16-31 K 16..31.
      const int ka = half * 192 + kk * 32 + (hi ? 16 : 0);
      AFrag a;
      a.h[0] = *(const v8bf*)(W + row * CIN + ka);
      a.h[1] = *(const v8bf*)(W + row * CIN + ka + 8);
      // B (32x16 bf16): lanes 0-15 hold K 0..15 of column `col`, lanes 16-31 K 16..31.
      const v16bf b = *(const v16bf*)(X + col * XPITCH + kk * 32 + (hi ? 16 : 0));
      c = __builtin_amdgcn_wmma_f32_16x16x32_bf16(false, a.v, false, b,
                                                  (short)0, c, false, false);
    }
#pragma unroll
    for (int r = 0; r < 8; ++r)
      D[(mt * 16 + hi * 8 + r) * DPITCH + col] = c[r];
  }
}

__global__ void wprep_kernel(const float* __restrict__ wp,
                             const float* __restrict__ wg,
                             __bf16* __restrict__ W) {
  const int i = blockIdx.x * 256 + threadIdx.x;
  if (i < CMB * CIN) {
    const int r = i / CIN, c = i - r * CIN;
    const float v = (r < 48) ? wp[r * CIN + c] : wg[(r - 48) * CIN + c];
    W[i] = (__bf16)v;
  }
}

__global__ __launch_bounds__(NTHREADS)
void policy_head_kernel(const float* __restrict__ x,
                        const float* __restrict__ mask,
                        const float* __restrict__ msum_in,
                        const __bf16* __restrict__ W,
                        const float* __restrict__ beta_g,
                        const float* __restrict__ wlg,
                        const float* __restrict__ wlp,
                        const float* __restrict__ blp,
                        const float* __restrict__ wlp2,
                        const float* __restrict__ beta2,
                        const float* __restrict__ wc2,
                        float* __restrict__ out) {
  extern __shared__ char smem[];
  float*  D     = (float*)(smem);            // [96][368] fp32 GEMM result
  __bf16* X     = (__bf16*)(smem + XOFF);    // [368 pos][208] bf16 x-half
  float*  gp    = (float*)(smem + GPOFF);    // gpooled[144]
  float*  add48 = gp + 144;                  // outg_lin + beta2 [48]
  float*  hid   = add48 + 48;                // pass hidden [48]
  float*  wc2s  = hid + 48;                  // staged conv2 weights [96]

  const int n    = blockIdx.x;
  const int tid  = threadIdx.x;
  const int wave = tid >> 5;
  const int lane = tid & 31;
  const float* xn   = x + (size_t)n * CIN * HW;
  const float* mkn  = mask + (size_t)n * HW;
  const float  msum = msum_in[n];

  stage_x(xn, X, tid, 0);
  __syncthreads();
  gemm_half(W, X, D, wave, lane, 0);
  __syncthreads();
  stage_x(xn, X, tid, 1);
  __syncthreads();
  gemm_half(W, X, D, wave, lane, 1);
  __syncthreads();

  // KataGPool over the g-branch (rows 48..95): masked relu mean + masked max.
  {
    const float inv = 1.0f / msum;
    int g = wave * 3;                     // 16 waves * 3 channels = 48
    for (int i = 0; i < 3; ++i, ++g) {
      const float* row = D + (48 + g) * DPITCH;
      const float  bg  = beta_g[g];
      float s = 0.0f, m = -1e30f;
      for (int pos = lane; pos < HW; pos += 32) {
        const float mk = mkn[pos];
        float a = (row[pos] + bg) * mk;
        a = a > 0.0f ? a : 0.0f;
        s += a;
        const float lm = a + mk - 1.0f;
        m = m > lm ? m : lm;
      }
#pragma unroll
      for (int off = 16; off > 0; off >>= 1) {   // wave32 reduction
        s += __shfl_xor(s, off, 32);
        const float om = __shfl_xor(m, off, 32);
        m = m > om ? m : om;
      }
      if (lane == 0) { gp[g] = s * inv; gp[96 + g] = m; }
    }
  }
  __syncthreads();
  if (tid < 48) {
    const float off10 = (sqrtf(msum) - 14.0f) * 0.1f;
    gp[48 + tid] = gp[tid] * off10;
  }
  __syncthreads();
  if (tid < 48) {                       // outg_lin (+beta2 folded in)
    float s = 0.0f;
    const float* wr = wlg + tid * 144;
    for (int j = 0; j < 144; ++j) s += gp[j] * wr[j];
    add48[tid] = s + beta2[tid];
  } else if (tid >= 64 && tid < 112) {  // pass hidden
    const int h = tid - 64;
    float s = blp[h];
    const float* wr = wlp + h * 144;
    for (int j = 0; j < 144; ++j) s += gp[j] * wr[j];
    hid[h] = s > 0.0f ? s : 0.0f;
  } else if (tid >= 128 && tid < 224) { // stage conv2 weights
    wc2s[tid - 128] = wc2[tid - 128];
  }
  __syncthreads();

  if (tid < 2) {                        // pass logits -> position L (=361)
    float s = 0.0f;
    const float* wr = wlp2 + tid * 48;
    for (int p = 0; p < 48; ++p) s += hid[p] * wr[p];
    out[(size_t)n * 2172 + (tid ? 5 : 0) * 362 + 361] = s;
  }

  // Fused: outp + broadcast add, relu*mask, 48->2 conv, mask penalty, pack.
  for (int pos = tid; pos < HW; pos += NTHREADS) {
    const float mk = mkn[pos];
    float a0 = 0.0f, a1 = 0.0f;
    for (int p = 0; p < 48; ++p) {
      float v = (D[p * DPITCH + pos] + add48[p]) * mk;
      v = v > 0.0f ? v : 0.0f;
      a0 += wc2s[p] * v;
      a1 += wc2s[48 + p] * v;
    }
    const float pen = (1.0f - mk) * 5000.0f;
    out[(size_t)n * 2172 + pos] = a0 - pen;
    out[(size_t)n * 2172 + 5 * 362 + pos] = a1 - pen;
  }
  for (int e = tid; e < 4 * 362; e += NTHREADS)   // zero channels 1..4
    out[(size_t)n * 2172 + 362 + e] = 0.0f;
}

extern "C" void kernel_launch(void* const* d_in, const int* in_sizes, int n_in,
                              void* d_out, int out_size, void* d_ws, size_t ws_size,
                              hipStream_t stream) {
  (void)in_sizes; (void)n_in; (void)out_size; (void)ws_size;
  const float* x      = (const float*)d_in[0];
  const float* mask   = (const float*)d_in[1];
  const float* msum   = (const float*)d_in[2];
  const float* wp     = (const float*)d_in[3];
  const float* wg     = (const float*)d_in[4];
  const float* beta_g = (const float*)d_in[5];
  const float* wlg    = (const float*)d_in[6];
  const float* wlp    = (const float*)d_in[7];
  const float* blp    = (const float*)d_in[8];
  const float* wlp2   = (const float*)d_in[9];
  const float* beta2  = (const float*)d_in[10];
  const float* wc2    = (const float*)d_in[11];
  float*  out = (float*)d_out;
  __bf16* W   = (__bf16*)d_ws;   // 96*384 bf16 = 73728 B of scratch

  wprep_kernel<<<(CMB * CIN + 255) / 256, 256, 0, stream>>>(wp, wg, W);

  (void)hipFuncSetAttribute((const void*)policy_head_kernel,
                            hipFuncAttributeMaxDynamicSharedMemorySize,
                            (int)SMEM_TOTAL);
  policy_head_kernel<<<1024, NTHREADS, SMEM_TOTAL, stream>>>(
      x, mask, msum, W, beta_g, wlg, wlp, blp, wlp2, beta2, wc2, out);
}
